// RnnMsgAgent_47717086658681
// MI455X (gfx1250) — matrix operand order repaired and checked by reading, not compiled, not run on hardware
//
#include <hip/hip_runtime.h>
#include <hip/hip_bf16.h>
#include <cstdint>
#include <cstddef>

// ---- problem dims (compile-time) ----
#define NBATCH   8192
#define NAGENT   16
#define HID      256
#define NK       32
#define NQ       32
#define NV       128
#define IN_DIM   512
#define CUR_HID  512
#define CUR_OUT  128
#define N_ACT    64
#define QK_DIM   (NQ + NK * NAGENT)   // 544
#define CAT_DIM  (NV + HID)           // 384

#define GB       2                    // batches per workgroup in kernel 1
#define ROWS     (NAGENT * GB)        // 32

typedef __attribute__((ext_vector_type(16))) _Float16 v16h;
typedef __attribute__((ext_vector_type(8)))  _Float16 v8h;
typedef __attribute__((ext_vector_type(8)))  float    v8f;

// ---------------------------------------------------------------------------
// WMMA fragment helpers (gfx1250 wave32, V_WMMA_F32_16X16X32_F16 layouts)
// A 16x32 f16: lane r (0..15) & lane r+16 both hold row M=r.
//   lane<16 : elems 0..15 -> K=k0+0..15 (contiguous, same order as B)
//   lane>=16: elems 0..15 -> K=k0+16..31
// B 32x16 f16: lane r -> column N=n0+r; elem e -> K=k0+e (+16 for lanes>=16)
//   With weights pre-transposed to f16 [N][K], a B fragment is 32 contiguous
//   bytes per lane: two global_load_b128.
// C/D 16x16 f32: vgpr i, lane r -> (M=i+8*hi, N=n0+r)
// ---------------------------------------------------------------------------

__device__ __forceinline__ v16h load_a_frag(const _Float16* __restrict__ X,
                                            int ldx, int r, int hi, int k0) {
    union { v16h v; v8h h[2]; } u;
    const _Float16* p = X + r * ldx + k0 + hi * 16;
    u.h[0] = *(const v8h*)(p);        // K = k0 + hi*16 + 0..7
    u.h[1] = *(const v8h*)(p + 8);    // K = k0 + hi*16 + 8..15
    return u.v;
}

// One 16x16 output tile at column n0 of X[16 x K] * W (Wt = f16 [N][K]).
__device__ __forceinline__ v8f gemm_tile(const _Float16* __restrict__ Xlds, int ldx,
                                         const _Float16* __restrict__ Wt, int K,
                                         int n0, int lane) {
    const int r = lane & 15, hi = lane >> 4;
    const _Float16* wp = Wt + (size_t)(n0 + r) * K + hi * 16;
    v8f acc = {};
    for (int k0 = 0; k0 < K; k0 += 32) {
        v16h a = load_a_frag(Xlds, ldx, r, hi, k0);
        union { v16h v; v8h h[2]; } ub;
        ub.h[0] = *(const v8h*)(wp + k0);      // K = k0 + hi*16 + 0..7
        ub.h[1] = *(const v8h*)(wp + k0 + 8);  // K = k0 + hi*16 + 8..15
        acc = __builtin_amdgcn_wmma_f32_16x16x32_f16(
            false, a, false, ub.v, (short)0, acc, false, false);
    }
    return acc;
}

// Two M-tiles (rows 0..15 and 16..31 of Xlds) sharing each B fragment:
// one B load pair feeds two WMMAs (halves weight traffic / VMEM issue).
__device__ __forceinline__ void gemm_tile_2m(const _Float16* __restrict__ Xlds, int ldx,
                                             const _Float16* __restrict__ Wt, int K,
                                             int n0, int lane, v8f& o0, v8f& o1) {
    const int r = lane & 15, hi = lane >> 4;
    const _Float16* wp = Wt + (size_t)(n0 + r) * K + hi * 16;
    v8f acc0 = {}, acc1 = {};
    for (int k0 = 0; k0 < K; k0 += 32) {
        union { v16h v; v8h h[2]; } ub;
        ub.h[0] = *(const v8h*)(wp + k0);
        ub.h[1] = *(const v8h*)(wp + k0 + 8);
        v16h a0 = load_a_frag(Xlds, ldx, r, hi, k0);
        v16h a1 = load_a_frag(Xlds + NAGENT * ldx, ldx, r, hi, k0);
        acc0 = __builtin_amdgcn_wmma_f32_16x16x32_f16(
            false, a0, false, ub.v, (short)0, acc0, false, false);
        acc1 = __builtin_amdgcn_wmma_f32_16x16x32_f16(
            false, a1, false, ub.v, (short)0, acc1, false, false);
    }
    o0 = acc0; o1 = acc1;
}

__device__ __forceinline__ void store_tile_h(v8f acc, _Float16* __restrict__ O, int ldo,
                                             int n0, const float* __restrict__ bias,
                                             int lane, bool do_relu) {
    const int r = lane & 15, hi = lane >> 4;
    const float bv = bias[n0 + r];
#pragma unroll
    for (int i = 0; i < 8; ++i) {
        float v = acc[i] + bv;
        if (do_relu) v = fmaxf(v, 0.f);
        O[(i + hi * 8) * ldo + n0 + r] = (_Float16)v;
    }
}

__device__ __forceinline__ void store_tile_f(v8f acc, float* __restrict__ O, int ldo,
                                             int n0, const float* __restrict__ bias,
                                             int lane) {
    const int r = lane & 15, hi = lane >> 4;
    const float bv = bias[n0 + r];
#pragma unroll
    for (int i = 0; i < 8; ++i)
        O[(i + hi * 8) * ldo + n0 + r] = acc[i] + bv;
}

// ---------------------------------------------------------------------------
// Kernel 0: one-time weight reformat  W[K x N] f32  ->  Wt[N x K] f16
// ---------------------------------------------------------------------------
__global__ __launch_bounds__(256) void wtrans_kernel(const float* __restrict__ src,
                                                     _Float16* __restrict__ dst,
                                                     int K, int N) {
    const int idx = blockIdx.x * 256 + threadIdx.x;
    if (idx < K * N) {
        const int k = idx / N, n = idx - k * N;
        dst[(size_t)n * K + k] = (_Float16)src[idx];
    }
}

// ---------------------------------------------------------------------------
// Kernel 1: fused target/predictor MLPs -> raw intrinsic importance.
// One workgroup (256 thr = 8 wave32) per GB=2 batches (32 rows, 2 M-tiles);
// all intermediates live in LDS (112 KB of the WGP's 320 KB).
// ---------------------------------------------------------------------------
__global__ __launch_bounds__(256) void mlp_imp_kernel(
    const float* __restrict__ x,
    const _Float16* __restrict__ Wt1, const float* __restrict__ bt1,
    const _Float16* __restrict__ Wt2, const float* __restrict__ bt2,
    const _Float16* __restrict__ Wp1, const float* __restrict__ bp1,
    const _Float16* __restrict__ Wp2, const float* __restrict__ bp2,
    const _Float16* __restrict__ Wp3, const float* __restrict__ bp3,
    float* __restrict__ imp_raw) {

    __shared__ __attribute__((aligned(16))) _Float16 x16[ROWS * IN_DIM];    // 32 KB
    __shared__ __attribute__((aligned(16))) _Float16 hA[ROWS * CUR_HID];    // 32 KB
    __shared__ __attribute__((aligned(16))) _Float16 hB[ROWS * CUR_HID];    // 32 KB
    __shared__ __attribute__((aligned(16))) float    tar[ROWS * CUR_OUT];   // 16 KB
    __shared__ float rowsum[ROWS];

    const int t = threadIdx.x;
    const int lane = t & 31, wave = t >> 5;
    const size_t row0 = (size_t)blockIdx.x * ROWS;   // 32 consecutive rows

    const float* xrow = x + row0 * IN_DIM;
    for (int i = t; i < ROWS * IN_DIM; i += 256) x16[i] = (_Float16)xrow[i];
    if (t < ROWS) rowsum[t] = 0.f;
    __syncthreads();

    // h_t = relu(x @ Wt1 + bt1)   [32 x 512]
    for (int nt = wave; nt < CUR_HID / 16; nt += 8) {
        v8f a0, a1;
        gemm_tile_2m(x16, IN_DIM, Wt1, IN_DIM, nt * 16, lane, a0, a1);
        store_tile_h(a0, hA,                     CUR_HID, nt * 16, bt1, lane, true);
        store_tile_h(a1, hA + NAGENT * CUR_HID,  CUR_HID, nt * 16, bt1, lane, true);
    }
    __syncthreads();

    // tar = h_t @ Wt2 + bt2       [32 x 128]
    {
        const int nt = wave;  // 8 tiles, 8 waves
        v8f a0, a1;
        gemm_tile_2m(hA, CUR_HID, Wt2, CUR_HID, nt * 16, lane, a0, a1);
        store_tile_f(a0, tar,                    CUR_OUT, nt * 16, bt2, lane);
        store_tile_f(a1, tar + NAGENT * CUR_OUT, CUR_OUT, nt * 16, bt2, lane);
    }
    __syncthreads();

    // p1 = relu(x @ Wp1 + bp1)    [32 x 512]
    for (int nt = wave; nt < CUR_HID / 16; nt += 8) {
        v8f a0, a1;
        gemm_tile_2m(x16, IN_DIM, Wp1, IN_DIM, nt * 16, lane, a0, a1);
        store_tile_h(a0, hB,                     CUR_HID, nt * 16, bp1, lane, true);
        store_tile_h(a1, hB + NAGENT * CUR_HID,  CUR_HID, nt * 16, bp1, lane, true);
    }
    __syncthreads();

    // p2 = relu(p1 @ Wp2 + bp2)   [32 x 512]  (hA reusable after tar)
    for (int nt = wave; nt < CUR_HID / 16; nt += 8) {
        v8f a0, a1;
        gemm_tile_2m(hB, CUR_HID, Wp2, CUR_HID, nt * 16, lane, a0, a1);
        store_tile_h(a0, hA,                     CUR_HID, nt * 16, bp2, lane, true);
        store_tile_h(a1, hA + NAGENT * CUR_HID,  CUR_HID, nt * 16, bp2, lane, true);
    }
    __syncthreads();

    // pred = p2 @ Wp3 + bp3; accumulate (tar - pred)^2 per row
    {
        const int nt = wave;  // 8 tiles, 8 waves
        v8f a0, a1;
        gemm_tile_2m(hA, CUR_HID, Wp3, CUR_HID, nt * 16, lane, a0, a1);
        const int r = lane & 15, hi = lane >> 4;
        const float bv = bp3[nt * 16 + r];
#pragma unroll
        for (int i = 0; i < 8; ++i) {
            const int m = i + hi * 8;
            float d0 = a0[i] + bv - tar[m * CUR_OUT + nt * 16 + r];
            float d1 = a1[i] + bv - tar[(m + NAGENT) * CUR_OUT + nt * 16 + r];
            atomicAdd(&rowsum[m], d0 * d0);
            atomicAdd(&rowsum[m + NAGENT], d1 * d1);
        }
    }
    __syncthreads();

    if (t < ROWS) imp_raw[row0 + t] = rowsum[t] * (1.f / (float)CUR_OUT);
}

// ---------------------------------------------------------------------------
// Kernel 2: communication (K/V/Q + attention) + importance mixing + heads.
// One workgroup per batch; whole batch (16 agents) lives in LDS.
// ---------------------------------------------------------------------------
__global__ __launch_bounds__(256) void attn_head_kernel(
    const float* __restrict__ hidden,
    const _Float16* __restrict__ Wk, const float* __restrict__ bk,
    const _Float16* __restrict__ Wv, const float* __restrict__ bv,
    const _Float16* __restrict__ Wq, const float* __restrict__ bq,
    const _Float16* __restrict__ Wa, const float* __restrict__ ba,
    const _Float16* __restrict__ Wc, const float* __restrict__ bc,
    const _Float16* __restrict__ W2, const float* __restrict__ b2,
    const float* __restrict__ imp_raw,
    float* __restrict__ out) {

    __shared__ __attribute__((aligned(16))) _Float16 hid16[NAGENT * HID];      //  8 KB
    __shared__ __attribute__((aligned(16))) _Float16 key16[NAGENT * NK];       //  1 KB
    __shared__ __attribute__((aligned(16))) float    valf[NAGENT * NV];        //  8 KB
    __shared__ __attribute__((aligned(16))) _Float16 qk16[NAGENT * QK_DIM];    // 17 KB
    __shared__ __attribute__((aligned(16))) float    attn[NAGENT * NAGENT];    //  1 KB
    __shared__ float intsm[NAGENT];
    __shared__ __attribute__((aligned(16))) _Float16 comb_in[NAGENT * CAT_DIM]; // 12 KB
    __shared__ __attribute__((aligned(16))) _Float16 comb16[NAGENT * HID];      //  8 KB

    const int t = threadIdx.x, b = blockIdx.x;
    const int lane = t & 31, wave = t >> 5;

    const float* hrow = hidden + (size_t)b * NAGENT * HID;
    for (int i = t; i < NAGENT * HID; i += 256) hid16[i] = (_Float16)hrow[i];
    __syncthreads();

    // 12 projection tiles: key(2) | value(8) | query(2), K = HID = 256
    for (int tile = wave; tile < 12; tile += 8) {
        if (tile < 2) {
            v8f acc = gemm_tile(hid16, HID, Wk, HID, tile * 16, lane);
            store_tile_h(acc, key16, NK, tile * 16, bk, lane, false);
        } else if (tile < 10) {
            const int nt = tile - 2;
            v8f acc = gemm_tile(hid16, HID, Wv, HID, nt * 16, lane);
            store_tile_f(acc, valf, NV, nt * 16, bv, lane);
        } else {
            const int nt = tile - 10;  // query -> qk cols 0..31
            v8f acc = gemm_tile(hid16, HID, Wq, HID, nt * 16, lane);
            store_tile_h(acc, qk16, QK_DIM, nt * 16, bq, lane, false);
        }
    }
    __syncthreads();

    // broadcast flat key into qk cols 32..543 (same for every row)
    for (int i = t; i < NAGENT * (NK * NAGENT); i += 256) {
        const int row = i >> 9, j = i & 511;
        qk16[row * QK_DIM + NQ + j] = key16[j];
    }
    __syncthreads();

    // logits = qk @ Wa + ba  (single 16x16 tile, K = 544) on wave 0
    if (wave == 0) {
        v8f acc = gemm_tile(qk16, QK_DIM, Wa, QK_DIM, 0, lane);
        store_tile_f(acc, attn, NAGENT, 0, ba, lane);
    }
    // concurrently: softmax of intrinsic importance over the 16 agents
    if (t == 32) {
        float v[NAGENT], mx = -1e30f;
#pragma unroll
        for (int j = 0; j < NAGENT; ++j) {
            v[j] = imp_raw[(size_t)b * NAGENT + j];
            mx = fmaxf(mx, v[j]);
        }
        float s = 0.f;
#pragma unroll
        for (int j = 0; j < NAGENT; ++j) { v[j] = expf(v[j] - mx); s += v[j]; }
        const float inv = 1.f / s;
#pragma unroll
        for (int j = 0; j < NAGENT; ++j) intsm[j] = v[j] * inv;
    }
    __syncthreads();

    // row softmax of logits, then attn = softmax + 0.5 * intsm[j]
    if (t < NAGENT) {
        float mx = -1e30f;
#pragma unroll
        for (int j = 0; j < NAGENT; ++j) mx = fmaxf(mx, attn[t * NAGENT + j]);
        float e[NAGENT], s = 0.f;
#pragma unroll
        for (int j = 0; j < NAGENT; ++j) {
            e[j] = expf(attn[t * NAGENT + j] - mx);
            s += e[j];
        }
        const float inv = 1.f / s;
#pragma unroll
        for (int j = 0; j < NAGENT; ++j)
            attn[t * NAGENT + j] = e[j] * inv + 0.5f * intsm[j];
    }
    __syncthreads();

    // applied = attn @ value (16x16x128, VALU f32) -> comb_in cols 0..127
    for (int i = t; i < NAGENT * NV; i += 256) {
        const int m = i >> 7, v = i & (NV - 1);
        float s = 0.f;
#pragma unroll
        for (int j = 0; j < NAGENT; ++j) s += attn[m * NAGENT + j] * valf[j * NV + v];
        comb_in[m * CAT_DIM + v] = (_Float16)s;
    }
    // concat hidden -> comb_in cols 128..383
    for (int i = t; i < NAGENT * HID; i += 256) {
        const int m = i >> 8, c = i & (HID - 1);
        comb_in[m * CAT_DIM + NV + c] = hid16[m * HID + c];
    }
    __syncthreads();

    // comb = relu(comb_in @ Wc + bc)  [16 x 256], K = 384
    for (int nt = wave; nt < HID / 16; nt += 8) {
        v8f acc = gemm_tile(comb_in, CAT_DIM, Wc, CAT_DIM, nt * 16, lane);
        store_tile_h(acc, comb16, HID, nt * 16, bc, lane, true);
    }
    __syncthreads();

    // out = comb @ W2 + b2  [16 x 64], K = 256, written straight to global
    if (wave < N_ACT / 16) {
        const int nt = wave;
        v8f acc = gemm_tile(comb16, HID, W2, HID, nt * 16, lane);
        const int r = lane & 15, hi = lane >> 4;
        const float bv = b2[nt * 16 + r];
#pragma unroll
        for (int i = 0; i < 8; ++i) {
            const int m = i + hi * 8;
            out[((size_t)b * NAGENT + m) * N_ACT + nt * 16 + r] = acc[i] + bv;
        }
    }
}

// ---------------------------------------------------------------------------
extern "C" void kernel_launch(void* const* d_in, const int* in_sizes, int n_in,
                              void* d_out, int out_size, void* d_ws, size_t ws_size,
                              hipStream_t stream) {
    (void)in_sizes; (void)n_in; (void)out_size; (void)ws_size;
    const float* x      = (const float*)d_in[0];
    const float* hidden = (const float*)d_in[1];
    const float* Wk  = (const float*)d_in[2];  const float* bk  = (const float*)d_in[3];
    const float* Wv  = (const float*)d_in[4];  const float* bv  = (const float*)d_in[5];
    const float* Wq  = (const float*)d_in[6];  const float* bq  = (const float*)d_in[7];
    const float* Wa  = (const float*)d_in[8];  const float* ba  = (const float*)d_in[9];
    const float* Wt1 = (const float*)d_in[10]; const float* bt1 = (const float*)d_in[11];
    const float* Wt2 = (const float*)d_in[12]; const float* bt2 = (const float*)d_in[13];
    const float* Wp1 = (const float*)d_in[14]; const float* bp1 = (const float*)d_in[15];
    const float* Wp2 = (const float*)d_in[16]; const float* bp2 = (const float*)d_in[17];
    const float* Wp3 = (const float*)d_in[18]; const float* bp3 = (const float*)d_in[19];
    const float* Wc  = (const float*)d_in[20]; const float* bc  = (const float*)d_in[21];
    const float* W2  = (const float*)d_in[22]; const float* b2  = (const float*)d_in[23];

    // ---- workspace layout ----
    float* imp_raw = (float*)d_ws;                        // 8192*16 f32 = 512 KB
    _Float16* wbase = (_Float16*)((char*)d_ws + (size_t)NBATCH * NAGENT * sizeof(float));
    size_t off = 0;
    _Float16* Wt1T = wbase + off; off += (size_t)IN_DIM  * CUR_HID;  // 512x512
    _Float16* Wt2T = wbase + off; off += (size_t)CUR_HID * CUR_OUT;  // 512x128
    _Float16* Wp1T = wbase + off; off += (size_t)IN_DIM  * CUR_HID;
    _Float16* Wp2T = wbase + off; off += (size_t)CUR_HID * CUR_HID;
    _Float16* Wp3T = wbase + off; off += (size_t)CUR_HID * CUR_OUT;
    _Float16* WkT  = wbase + off; off += (size_t)HID * NK;
    _Float16* WvT  = wbase + off; off += (size_t)HID * NV;
    _Float16* WqT  = wbase + off; off += (size_t)HID * NQ;
    _Float16* WaT  = wbase + off; off += (size_t)QK_DIM * NAGENT;
    _Float16* WcT  = wbase + off; off += (size_t)CAT_DIM * HID;
    _Float16* W2T  = wbase + off; off += (size_t)HID * N_ACT;

    auto xpose = [&](const float* s, _Float16* d, int K, int N) {
        const int n = K * N;
        wtrans_kernel<<<(n + 255) / 256, 256, 0, stream>>>(s, d, K, N);
    };
    xpose(Wt1, Wt1T, IN_DIM,  CUR_HID);
    xpose(Wt2, Wt2T, CUR_HID, CUR_OUT);
    xpose(Wp1, Wp1T, IN_DIM,  CUR_HID);
    xpose(Wp2, Wp2T, CUR_HID, CUR_HID);
    xpose(Wp3, Wp3T, CUR_HID, CUR_OUT);
    xpose(Wk,  WkT,  HID, NK);
    xpose(Wv,  WvT,  HID, NV);
    xpose(Wq,  WqT,  HID, NQ);
    xpose(Wa,  WaT,  QK_DIM, NAGENT);
    xpose(Wc,  WcT,  CAT_DIM, HID);
    xpose(W2,  W2T,  HID, N_ACT);

    mlp_imp_kernel<<<NBATCH / GB, 256, 0, stream>>>(
        x, Wt1T, bt1, Wt2T, bt2, Wp1T, bp1, Wp2T, bp2, Wp3T, bp3, imp_raw);

    attn_head_kernel<<<NBATCH, 256, 0, stream>>>(
        hidden, WkT, bk, WvT, bv, WqT, bq, WaT, ba, WcT, bc, W2T, b2,
        imp_raw, (float*)d_out);
}
